// MultiheadAttention_2980707303459
// MI455X (gfx1250) — matrix operand order repaired and checked by reading, not compiled, not run on hardware
//
#include <hip/hip_runtime.h>
#include <hip/hip_bf16.h>

// ---------------------------------------------------------------------------
// MultiheadAttention forward for MI455X (gfx1250, wave32, WMMA bf16 + TDM)
// B=1, N=4096, D=1024, H=16, E=64
// ---------------------------------------------------------------------------

typedef __bf16 bf16_t;
typedef __attribute__((ext_vector_type(16))) __bf16 v16bf;
typedef __attribute__((ext_vector_type(8)))  __bf16 bf16x8;
typedef __attribute__((ext_vector_type(8)))  float  v8f;
typedef __attribute__((ext_vector_type(4)))  unsigned int u32x4;
typedef __attribute__((ext_vector_type(8)))  int  i32x8;
typedef __attribute__((ext_vector_type(4)))  int  i32x4;

#define N_TOK   4096
#define D_MODEL 1024
#define N_HEAD  16
#define E_HEAD  64

static __device__ __forceinline__ int lane_id() { return (int)(threadIdx.x & 31u); }

static __device__ __forceinline__ v16bf cat8(bf16x8 lo, bf16x8 hi) {
  return __builtin_shufflevector(lo, hi, 0,1,2,3,4,5,6,7,8,9,10,11,12,13,14,15);
}

// A-matrix 16x32 bf16 fragment. p -> element [m=0][k=0], ld = row stride (elems).
// ISA layout: lanes 0-15 row m hold K=0..7 / K=16..23; lanes 16-31 K=8..15 / 24..31.
static __device__ __forceinline__ v16bf load_a_frag(const bf16_t* __restrict__ p, int ld) {
  const int l = lane_id();
  const bf16_t* r = p + (size_t)(l & 15) * ld + ((l & 16) ? 8 : 0);
  bf16x8 lo = *(const bf16x8*)(r);       // K = off + 0..7
  bf16x8 hi = *(const bf16x8*)(r + 16);  // K = off + 16..23
  return cat8(lo, hi);
}

// B-matrix 32x16 bf16 fragment, source stored K-contiguous per column.
// ISA layout: lanes 0-15 (col n) hold K=0..7 / K=16..23; lanes 16-31 K=8..15 / 24..31.
static __device__ __forceinline__ v16bf load_b_frag(const bf16_t* __restrict__ p, int ld) {
  const int l = lane_id();
  const bf16_t* r = p + (size_t)(l & 15) * ld + ((l & 16) ? 8 : 0);
  bf16x8 lo = *(const bf16x8*)(r);
  bf16x8 hi = *(const bf16x8*)(r + 16);
  return cat8(lo, hi);
}

static __device__ __forceinline__ v8f wmma_bf16(v16bf a, v16bf b, v8f c) {
  // (neg_a, A, neg_b, B, c_mod, C, reuse_a, reuse_b)
  return __builtin_amdgcn_wmma_f32_16x16x32_bf16(false, a, false, b, (short)0, c,
                                                 false, false);
}

// ---------------------------------------------------------------------------
// Tensor Data Mover: 2D tile load (rows x 32 bf16) global -> LDS with padding.
// D# built per CDNA5 ISA 8.3/8.4: count=1, type=2, data_size=1 (2B),
// pad_enable=1, pad_interval=3 (16 DWORDs = one 64B row), pad_amount=3 (4 DW)
// => LDS row stride 80B (40 bf16), conflict-free for ds_load_b128 fragments.
// ---------------------------------------------------------------------------
static __device__ __forceinline__ void tdm_load_tile_128x32(
    const bf16_t* gptr, unsigned int lds_byte_off, int row_stride_elems) {
  unsigned long long ga = (unsigned long long)(uintptr_t)gptr;
  u32x4 g0;
  g0[0] = 1u;                                   // count=1 (valid), user mode
  g0[1] = lds_byte_off;                         // lds_addr
  g0[2] = (unsigned int)ga;                     // global_addr[31:0]
  g0[3] = ((unsigned int)(ga >> 32) & 0x01FFFFFFu) | (2u << 30);  // ga[56:32] | type=2

  i32x8 g1;
  g1[0] = (int)((1u << 16) |                    // data_size = 1 -> 2 bytes
                (1u << 20) |                    // pad_enable
                (3u << 22) |                    // pad_interval: 16 DWORDs
                (3u << 25));                    // pad_amount:   4 DWORDs
  g1[1] = (int)(32u << 16);                     // tensor_dim0[15:0] = 32
  g1[2] = (int)(128u << 16);                    // tensor_dim1[15:0] = 128
  g1[3] = (int)(32u << 16);                     // tile_dim0 = 32
  g1[4] = (int)(128u);                          // tile_dim1 = 128, tile_dim2 = 0
  g1[5] = (int)(unsigned int)row_stride_elems;  // tensor_dim0_stride[31:0]
  g1[6] = 0;
  g1[7] = 0;

  i32x4 z4 = {0, 0, 0, 0};
#if __has_include(<hip/amd_detail/amd_gfx1250_TDM.h>)
  i32x8 z8 = {0, 0, 0, 0, 0, 0, 0, 0};
  __builtin_amdgcn_tensor_load_to_lds(g0, g1, z4, z4, z8, 0);
#else
  __builtin_amdgcn_tensor_load_to_lds(g0, g1, z4, z4, 0);
#endif
}

// Cross-lane reductions within 16-lane column groups (C-layout rows).
static __device__ __forceinline__ v8f reduce_max16(v8f t) {
#pragma unroll
  for (int off = 1; off < 16; off <<= 1) {
#pragma unroll
    for (int i = 0; i < 8; ++i) {
      float u = __shfl_xor(t[i], off, 32);
      t[i] = fmaxf(t[i], u);
    }
  }
  return t;
}

static __device__ __forceinline__ v8f reduce_sum16(v8f t) {
#pragma unroll
  for (int off = 1; off < 16; off <<= 1) {
#pragma unroll
    for (int i = 0; i < 8; ++i) {
      float u = __shfl_xor(t[i], off, 32);
      t[i] = t[i] + u;
    }
  }
  return t;
}

// ---------------------------------------------------------------------------
// fp32 -> bf16 conversion
// ---------------------------------------------------------------------------
__global__ void cvt_f32_bf16(const float* __restrict__ s, bf16_t* __restrict__ d, int n) {
  int i = blockIdx.x * blockDim.x + threadIdx.x;
  int stride = gridDim.x * blockDim.x;
  for (; i < n; i += stride) d[i] = (bf16_t)s[i];
}

// ---------------------------------------------------------------------------
// bf16 transpose: d[c][r] = s[r][c]   (s is rows x cols)
// ---------------------------------------------------------------------------
__global__ void transpose_bf16(const bf16_t* __restrict__ s, bf16_t* __restrict__ d,
                               int rows, int cols) {
  size_t i = (size_t)blockIdx.x * blockDim.x + threadIdx.x;
  size_t n = (size_t)rows * cols;
  size_t stride = (size_t)gridDim.x * blockDim.x;
  for (; i < n; i += stride) {
    int r = (int)(i / cols);
    int c = (int)(i % cols);
    d[(size_t)c * rows + r] = s[i];
  }
}

// ---------------------------------------------------------------------------
// C = A[M,K] @ Bt[N,K]^T  (bf16 in, fp32 accumulate), TDM-staged LDS tiles.
// Block: 256 threads = 8 waves; block tile 128(M) x 128(N); wave tile 32x64.
// Wave 0 streams double-buffered 128x32 A/B tiles via tensor_load_to_lds,
// pipelined one k-step ahead (s_wait_tensorcnt 2).
// If Cf != nullptr: fp32 output + bias, else bf16 output to Cb.
// ---------------------------------------------------------------------------
__global__ void __launch_bounds__(256) gemm_bf16_nt(
    const bf16_t* __restrict__ A, const bf16_t* __restrict__ Bt,
    bf16_t* __restrict__ Cb, float* __restrict__ Cf, const float* __restrict__ bias,
    int M, int Nn, int K) {
  __shared__ bf16_t As[2][128][40];  // 80B row stride (TDM pad)
  __shared__ bf16_t Bs[2][128][40];

  const int wave = (int)(threadIdx.x >> 5);
  const int l = lane_id();
  const int wm = wave & 3;   // 4 waves along M (32 rows each)
  const int wn = wave >> 2;  // 2 waves along N (64 cols each)
  const int m0 = blockIdx.y * 128;
  const int n0 = blockIdx.x * 128;

  const v8f z = {0.f, 0.f, 0.f, 0.f, 0.f, 0.f, 0.f, 0.f};
  v8f c[2][4];
#pragma unroll
  for (int rf = 0; rf < 2; ++rf)
#pragma unroll
    for (int cf = 0; cf < 4; ++cf) c[rf][cf] = z;

  const int nkt = K / 32;
  if (wave == 0) {
    tdm_load_tile_128x32(A + (size_t)m0 * K,
                         (unsigned int)(uintptr_t)&As[0][0][0], K);
    tdm_load_tile_128x32(Bt + (size_t)n0 * K,
                         (unsigned int)(uintptr_t)&Bs[0][0][0], K);
  }

  for (int kt = 0; kt < nkt; ++kt) {
    const int p = kt & 1;
    if (wave == 0) {
      if (kt + 1 < nkt) {  // prefetch next tiles, then wait for current pair
        tdm_load_tile_128x32(A + (size_t)m0 * K + (size_t)(kt + 1) * 32,
                             (unsigned int)(uintptr_t)&As[1 - p][0][0], K);
        tdm_load_tile_128x32(Bt + (size_t)n0 * K + (size_t)(kt + 1) * 32,
                             (unsigned int)(uintptr_t)&Bs[1 - p][0][0], K);
        __builtin_amdgcn_s_wait_tensorcnt(2);
      } else {
        __builtin_amdgcn_s_wait_tensorcnt(0);
      }
    }
    __syncthreads();  // publish tile p to all waves

    v16bf a[2], b[4];
#pragma unroll
    for (int rf = 0; rf < 2; ++rf)
      a[rf] = load_a_frag(&As[p][wm * 32 + rf * 16][0], 40);
#pragma unroll
    for (int cf = 0; cf < 4; ++cf)
      b[cf] = load_b_frag(&Bs[p][wn * 64 + cf * 16][0], 40);
#pragma unroll
    for (int rf = 0; rf < 2; ++rf)
#pragma unroll
      for (int cf = 0; cf < 4; ++cf)
        c[rf][cf] = wmma_bf16(a[rf], b[cf], c[rf][cf]);

    __syncthreads();  // all waves done reading buffer p before it is refilled
  }

  const int col = l & 15;
  const int ro = (l & 16) ? 8 : 0;
#pragma unroll
  for (int rf = 0; rf < 2; ++rf) {
#pragma unroll
    for (int cf = 0; cf < 4; ++cf) {
      const int cc = n0 + wn * 64 + cf * 16 + col;
      const int rr = m0 + wm * 32 + rf * 16 + ro;
      if (Cf != nullptr) {
        const float bb = (bias != nullptr) ? bias[cc] : 0.f;
#pragma unroll
        for (int i = 0; i < 8; ++i)
          Cf[(size_t)(rr + i) * Nn + cc] = c[rf][cf][i] + bb;
      } else {
#pragma unroll
        for (int i = 0; i < 8; ++i)
          Cb[(size_t)(rr + i) * Nn + cc] = (bf16_t)c[rf][cf][i];
      }
    }
  }
}

// ---------------------------------------------------------------------------
// Flash-attention forward. One wave owns 32 query rows of one head and scans
// all 4096 keys in tiles of 32. Block = 128 threads (4 waves).
// Q:[N][D] bf16, K:[N][D] bf16, Vt:[D][N] bf16, O:[N][D] bf16.
// ---------------------------------------------------------------------------
__global__ void __launch_bounds__(128) attn_fwd(
    const bf16_t* __restrict__ Q, const bf16_t* __restrict__ Kk,
    const bf16_t* __restrict__ Vt, bf16_t* __restrict__ O) {
  __shared__ bf16_t Pl[4][32][40];  // per-wave P tile, padded row stride

  const int wave = (int)(threadIdx.x >> 5);
  const int l = lane_id();
  const int h = blockIdx.y;
  const int q0 = blockIdx.x * 128 + wave * 32;
  const int hoff = h * E_HEAD;

  // Q fragments stay resident: 2 row-frags x 2 k-steps over E=64
  v16bf qf[2][2];
#pragma unroll
  for (int rf = 0; rf < 2; ++rf)
#pragma unroll
    for (int ks = 0; ks < 2; ++ks)
      qf[rf][ks] = load_a_frag(Q + (size_t)(q0 + rf * 16) * D_MODEL + hoff + ks * 32,
                               D_MODEL);

  const v8f z = {0.f, 0.f, 0.f, 0.f, 0.f, 0.f, 0.f, 0.f};
  v8f o[2][4];
  v8f mrow[2], lrow[2];
#pragma unroll
  for (int rf = 0; rf < 2; ++rf) {
#pragma unroll
    for (int cf = 0; cf < 4; ++cf) o[rf][cf] = z;
    lrow[rf] = z;
#pragma unroll
    for (int i = 0; i < 8; ++i) mrow[rf][i] = -3.0e38f;
  }

  const int col = l & 15;
  const int ro = (l & 16) ? 8 : 0;

  for (int kt = 0; kt < N_TOK / 32; ++kt) {
    if (kt + 4 < N_TOK / 32) {  // warm L0/L2 a few tiles ahead
      __builtin_prefetch(Kk + (size_t)((kt + 4) * 32) * D_MODEL + hoff, 0, 3);
      __builtin_prefetch(Vt + (size_t)hoff * N_TOK + (kt + 4) * 32, 0, 3);
    }

    // ---- S = (Q K^T) * scale, tile 32(q) x 32(keys), contraction E=64 ----
    v8f s[2][2];
#pragma unroll
    for (int rf = 0; rf < 2; ++rf)
#pragma unroll
      for (int cf = 0; cf < 2; ++cf) s[rf][cf] = z;

#pragma unroll
    for (int cf = 0; cf < 2; ++cf) {
      const bf16_t* kb = Kk + (size_t)(kt * 32 + cf * 16) * D_MODEL + hoff;
      v16bf k0 = load_b_frag(kb, D_MODEL);
      v16bf k1 = load_b_frag(kb + 32, D_MODEL);
#pragma unroll
      for (int rf = 0; rf < 2; ++rf) {
        s[rf][cf] = wmma_bf16(qf[rf][0], k0, s[rf][cf]);
        s[rf][cf] = wmma_bf16(qf[rf][1], k1, s[rf][cf]);
      }
    }

    // ---- online softmax (fp32) ----
#pragma unroll
    for (int rf = 0; rf < 2; ++rf) {
      v8f t;
#pragma unroll
      for (int i = 0; i < 8; ++i) {
        s[rf][0][i] = s[rf][0][i] * 0.125f;  // 1/sqrt(64)
        s[rf][1][i] = s[rf][1][i] * 0.125f;
        t[i] = fmaxf(s[rf][0][i], s[rf][1][i]);
      }
      t = reduce_max16(t);
      v8f mnew, alpha, p0, p1, rs;
#pragma unroll
      for (int i = 0; i < 8; ++i) {
        mnew[i] = fmaxf(mrow[rf][i], t[i]);
        alpha[i] = __expf(mrow[rf][i] - mnew[i]);
        p0[i] = __expf(s[rf][0][i] - mnew[i]);
        p1[i] = __expf(s[rf][1][i] - mnew[i]);
        rs[i] = p0[i] + p1[i];
      }
      rs = reduce_sum16(rs);
#pragma unroll
      for (int i = 0; i < 8; ++i) {
        lrow[rf][i] = lrow[rf][i] * alpha[i] + rs[i];
        mrow[rf][i] = mnew[i];
      }
#pragma unroll
      for (int cf = 0; cf < 4; ++cf)
#pragma unroll
        for (int i = 0; i < 8; ++i) o[rf][cf][i] = o[rf][cf][i] * alpha[i];

      // relayout P: C-layout regs -> LDS row-major [qrow][key]
#pragma unroll
      for (int i = 0; i < 8; ++i) {
        Pl[wave][rf * 16 + ro + i][col] = (bf16_t)p0[i];
        Pl[wave][rf * 16 + ro + i][16 + col] = (bf16_t)p1[i];
      }
    }

    // ---- O += P V  (contraction over 32 keys; same-wave LDS, in-order) ----
    v16bf pf[2];
    pf[0] = load_a_frag(&Pl[wave][0][0], 40);
    pf[1] = load_a_frag(&Pl[wave][16][0], 40);
#pragma unroll
    for (int cf = 0; cf < 4; ++cf) {
      v16bf vf = load_b_frag(Vt + (size_t)(hoff + cf * 16) * N_TOK + kt * 32, N_TOK);
      o[0][cf] = wmma_bf16(pf[0], vf, o[0][cf]);
      o[1][cf] = wmma_bf16(pf[1], vf, o[1][cf]);
    }
  }

  // ---- epilogue: O /= l, store bf16 at [token][h*64 + e] ----
#pragma unroll
  for (int rf = 0; rf < 2; ++rf) {
#pragma unroll
    for (int cf = 0; cf < 4; ++cf) {
      bf16_t* p = O + (size_t)(q0 + rf * 16) * D_MODEL + hoff + cf * 16 + col;
#pragma unroll
      for (int i = 0; i < 8; ++i) {
        float inv = 1.0f / lrow[rf][i];
        p[(size_t)(ro + i) * D_MODEL] = (bf16_t)(o[rf][cf][i] * inv);
      }
    }
  }
}

// ---------------------------------------------------------------------------
// Host launcher
// ---------------------------------------------------------------------------
extern "C" void kernel_launch(void* const* d_in, const int* in_sizes, int n_in,
                              void* d_out, int out_size, void* d_ws, size_t ws_size,
                              hipStream_t stream) {
  (void)in_sizes; (void)n_in; (void)out_size; (void)ws_size;
  const float* query = (const float*)d_in[0];
  const float* Wq = (const float*)d_in[1];
  const float* Wk = (const float*)d_in[2];
  const float* Wv = (const float*)d_in[3];
  const float* Wo = (const float*)d_in[4];
  const float* bo = (const float*)d_in[5];
  float* out = (float*)d_out;

  const size_t ND = (size_t)N_TOK * D_MODEL;   // 4M elems
  const size_t DD = (size_t)D_MODEL * D_MODEL; // 1M elems
  bf16_t* Xb  = (bf16_t*)d_ws;   // query bf16
  bf16_t* Wqb = Xb + ND;
  bf16_t* Wkb = Wqb + DD;
  bf16_t* Wvb = Wkb + DD;
  bf16_t* Wob = Wvb + DD;
  bf16_t* Qb  = Wob + DD;
  bf16_t* Kb  = Qb + ND;
  bf16_t* Vb  = Kb + ND;
  bf16_t* Vtb = Vb + ND;         // V transposed: [D][N]
  bf16_t* Ob  = Vtb + ND;        // attention output (head-concat) bf16

  cvt_f32_bf16<<<2048, 256, 0, stream>>>(query, Xb, (int)ND);
  cvt_f32_bf16<<<1024, 256, 0, stream>>>(Wq, Wqb, (int)DD);
  cvt_f32_bf16<<<1024, 256, 0, stream>>>(Wk, Wkb, (int)DD);
  cvt_f32_bf16<<<1024, 256, 0, stream>>>(Wv, Wvb, (int)DD);
  cvt_f32_bf16<<<1024, 256, 0, stream>>>(Wo, Wob, (int)DD);

  dim3 gg(D_MODEL / 128, N_TOK / 128);  // (8, 32), 256 threads each
  gemm_bf16_nt<<<gg, 256, 0, stream>>>(Xb, Wqb, Qb, nullptr, nullptr,
                                       N_TOK, D_MODEL, D_MODEL);
  gemm_bf16_nt<<<gg, 256, 0, stream>>>(Xb, Wkb, Kb, nullptr, nullptr,
                                       N_TOK, D_MODEL, D_MODEL);
  gemm_bf16_nt<<<gg, 256, 0, stream>>>(Xb, Wvb, Vb, nullptr, nullptr,
                                       N_TOK, D_MODEL, D_MODEL);

  transpose_bf16<<<2048, 256, 0, stream>>>(Vb, Vtb, N_TOK, D_MODEL);

  attn_fwd<<<dim3(N_TOK / 128, N_HEAD), 128, 0, stream>>>(Qb, Kb, Vtb, Ob);

  gemm_bf16_nt<<<gg, 256, 0, stream>>>(Ob, Wob, nullptr, out, bo,
                                       N_TOK, D_MODEL, D_MODEL);
}
